// Multihead_multifeats_QMixer_25503515803786
// MI455X (gfx1250) — matrix operand order repaired and checked by reading, not compiled, not run on hardware
//
#include <hip/hip_runtime.h>
#include <hip/hip_bf16.h>

// ============================================================================
// Multihead_multifeats_QMixer for MI455X (gfx1250, wave32, WMMA)
//
// Strategy (see analysis):
//  * p==2 of the hyper-network is dead in the reference -> skipped entirely.
//  * Big GEMMs in bf16 WMMA (v_wmma_f32_16x16x32_bf16), f32 accumulate.
//  * GEMM1 stacks ALL st-projections into one (B x 256) @ (256 x 768) GEMM:
//      n in [  0,512): h  = relu(st@hyp_w1^T)  (f=0..3, p=0..1) -> bf16
//      n in [512,576): z1 = relu(st@hw1_a^T)
//      n in [576,640): zf = relu(st@hwf_a^T)
//      n in [640,672): b1 = st@hb1_w^T         (no relu)
//      n in [672,704): v1 = relu(st@v1_w^T)
//      n in [704,736): fa = relu(st@fa_w1^T)
//      n in [736,768): zero pad
//  * GEMM2 (h @ hyp_w2^T, K=64, N=512 per (f,p)) is fused with the whole
//    attention + mixing + hypernet tail in one kernel; the (B x 4096) f32
//    W tensor never touches HBM (16x512 tile lives in LDS).
// ============================================================================

#define BATCH 16384   // BS*T
#define N1    768     // stacked GEMM1 output width (46 used tiles + pad)

typedef __attribute__((ext_vector_type(16))) __bf16 v16bf;
typedef __attribute__((ext_vector_type(8)))  float  v8f;

union BFrag { uint4 u[2]; v16bf v; };

__device__ __forceinline__ unsigned short f2bf(float x) {
  union { float f; unsigned u; } c; c.f = x;
  unsigned r;
  if ((c.u & 0x7f800000u) == 0x7f800000u) r = c.u;        // inf / nan
  else r = c.u + 0x7fffu + ((c.u >> 16) & 1u);            // round-nearest-even
  return (unsigned short)(r >> 16);
}

// ---------------------------------------------------------------------------
// Pack kernels
// ---------------------------------------------------------------------------
__global__ __launch_bounds__(256) void k_cvt_states(const float* __restrict__ st,
                                                    unsigned short* __restrict__ stbf) {
  int idx = (blockIdx.x * 256 + threadIdx.x) * 4;
  float4 v = *(const float4*)(st + idx);
  ushort4 o;
  o.x = f2bf(v.x); o.y = f2bf(v.y); o.z = f2bf(v.z); o.w = f2bf(v.w);
  *(ushort4*)(stbf + idx) = o;
}

__global__ __launch_bounds__(256) void k_pack_w1(
    const float* __restrict__ hyp_w1, const float* __restrict__ hyp_b1,
    const float* __restrict__ hw1_a,  const float* __restrict__ hw1_ab,
    const float* __restrict__ hwf_a,  const float* __restrict__ hwf_ab,
    const float* __restrict__ hb1_w,  const float* __restrict__ hb1_b,
    const float* __restrict__ v1_w,   const float* __restrict__ v1_b,
    const float* __restrict__ fa_w1,  const float* __restrict__ fa_b1,
    unsigned short* __restrict__ wp1, float* __restrict__ bp1) {
  int r = blockIdx.x, t = threadIdx.x;
  const float* src = nullptr; const float* bsrc = nullptr; int bi = 0;
  if (r < 512) {                       // hyper-w1, only p = 0,1 (p=2 is dead)
    int fp = r >> 6, k = r & 63, f = fp >> 1, p = fp & 1;
    src = hyp_w1 + (size_t)((f * 3 + p) * 64 + k) * 256;
    bsrc = hyp_b1; bi = (f * 3 + p) * 64 + k;
  } else if (r < 576) { src = hw1_a + (size_t)(r - 512) * 256; bsrc = hw1_ab; bi = r - 512; }
  else if (r < 640)   { src = hwf_a + (size_t)(r - 576) * 256; bsrc = hwf_ab; bi = r - 576; }
  else if (r < 672)   { src = hb1_w + (size_t)(r - 640) * 256; bsrc = hb1_b;  bi = r - 640; }
  else if (r < 704)   { src = v1_w  + (size_t)(r - 672) * 256; bsrc = v1_b;   bi = r - 672; }
  else if (r < 736)   { src = fa_w1 + (size_t)(r - 704) * 256; bsrc = fa_b1;  bi = r - 704; }
  wp1[(size_t)r * 256 + t] = src ? f2bf(src[t]) : (unsigned short)0;
  if (t == 0) bp1[r] = bsrc ? bsrc[bi] : 0.f;
}

__global__ __launch_bounds__(256) void k_pack_w2(const float* __restrict__ hyp_w2,
                                                 unsigned short* __restrict__ wp2) {
  int idx = blockIdx.x * 256 + threadIdx.x;    // [fp(8)][o(512)][k(64)]
  int fp = idx >> 15, rem = idx & 32767;
  int f = fp >> 1, p = fp & 1;
  wp2[idx] = f2bf(hyp_w2[(((size_t)(f * 3 + p)) << 15) + rem]);
}

// ---------------------------------------------------------------------------
// GEMM1: (B x 256) @ (256 x 768) in bf16 WMMA, f32 accumulate.
// grid = (B/16, 768/128); 8 waves/WG, each wave owns one 16x16 N-tile.
// ---------------------------------------------------------------------------
__global__ __launch_bounds__(256) void k_gemm1(
    const unsigned short* __restrict__ stbf,   // [B][256] bf16
    const unsigned short* __restrict__ wp1,    // [768][256] bf16 (N-major, K contig)
    const float* __restrict__ bp1,             // [768]
    unsigned short* __restrict__ hbuf,         // [B][512] bf16
    float* __restrict__ z1g, float* __restrict__ zfg,
    float* __restrict__ b1g, float* __restrict__ v1g, float* __restrict__ fa1g) {
  __shared__ __align__(16) unsigned short As[16 * 256];  // 8 KB A tile
  int t = threadIdx.x;
  int mt = blockIdx.x, ng = blockIdx.y;
  {
    int r = t >> 4, c = (t & 15) * 16;
    const uint4* src = (const uint4*)(stbf + (size_t)(mt * 16 + r) * 256 + c);
    uint4* dst = (uint4*)(As + r * 256 + c);
    dst[0] = src[0]; dst[1] = src[1];
  }
  __syncthreads();
  int wave = t >> 5, lane = t & 31, lm = lane & 15, kh = lane >> 4;
  int n0 = (ng * 8 + wave) * 16;
  const unsigned short* brow = wp1 + (size_t)(n0 + lm) * 256;
  v8f acc = {0.f, 0.f, 0.f, 0.f, 0.f, 0.f, 0.f, 0.f};
#pragma unroll
  for (int kk = 0; kk < 8; ++kk) {
    int K0 = kk * 32;
    BFrag a, b;
    // A: lane = row m, khalf-interleaved K (ISA 16-bit A-matrix layout)
    a.u[0] = *(const uint4*)(As + lm * 256 + K0 + kh * 8);
    a.u[1] = *(const uint4*)(As + lm * 256 + K0 + 16 + kh * 8);
    // B: lane = col n, khalf-interleaved K (same order as A)
    b.u[0] = *(const uint4*)(brow + K0 + kh * 8);
    b.u[1] = *(const uint4*)(brow + K0 + 16 + kh * 8);
    acc = __builtin_amdgcn_wmma_f32_16x16x32_bf16(false, a.v, false, b.v,
                                                  (short)0, acc, false, false);
  }
  int n = n0 + lm;
  float bias = bp1[n];
  size_t rb = (size_t)mt * 16;
#pragma unroll
  for (int r = 0; r < 8; ++r) {                 // C layout: m = r + 8*kh, n = lane&15
    size_t row = rb + r + 8 * kh;
    float v = acc[r] + bias;
    if (n < 512)      hbuf[row * 512 + n]       = f2bf(fmaxf(v, 0.f));
    else if (n < 576) z1g[row * 64 + (n - 512)] = fmaxf(v, 0.f);
    else if (n < 640) zfg[row * 64 + (n - 576)] = fmaxf(v, 0.f);
    else if (n < 672) b1g[row * 32 + (n - 640)] = v;
    else if (n < 704) v1g[row * 32 + (n - 672)] = fmaxf(v, 0.f);
    else if (n < 736) fa1g[row * 32 + (n - 704)] = fmaxf(v, 0.f);
  }
}

// ---------------------------------------------------------------------------
// Fused GEMM2 + attention + mixing + hypernet tail. One WG per 16 rows.
// ---------------------------------------------------------------------------
__global__ __launch_bounds__(256) void k_attn_mix_tail(
    const unsigned short* __restrict__ hbuf,  // [B][512] bf16
    const unsigned short* __restrict__ wp2,   // [8][512][64] bf16
    const float* __restrict__ hyp_b2,         // (4,3,512)
    const float* __restrict__ inproj_w,       // (4,3,8,8)
    const float* __restrict__ inproj_b,       // (4,3,8)
    const float* __restrict__ feats0, const float* __restrict__ feats1,
    const float* __restrict__ feats2, const float* __restrict__ feats3,
    const float* __restrict__ agent_qs,       // (B,8)
    const float* __restrict__ fa_w2, const float* __restrict__ fa_b2,
    const float* __restrict__ hw1_b, const float* __restrict__ hw1_bb,
    const float* __restrict__ hwf_b, const float* __restrict__ hwf_bb,
    const float* __restrict__ v2_w,  const float* __restrict__ v2_b,
    const float* __restrict__ z1g,   const float* __restrict__ zfg,
    const float* __restrict__ b1g,   const float* __restrict__ v1g,
    const float* __restrict__ fa1g,  float* __restrict__ out) {
  __shared__ __align__(16) float Wt[16][516];     // GEMM2 output tile (16x512)
  __shared__ __align__(16) float featL[16][516];  // feats for current f (16x512)
  __shared__ float vecL[16][64];
  __shared__ float qL[16][64];
  __shared__ float kL[16][64];
  __shared__ float attw[16][64];
  __shared__ float z1L[16][64];
  __shared__ float zfL[16][64];
  __shared__ float b1L[16][32];
  __shared__ float v1L[16][32];
  __shared__ float fa1L[16][32];
  __shared__ float fwL[16][4];
  __shared__ float mixedL[16][8];
  __shared__ float hidL[16][32];
  __shared__ float wfL[16][32];
  __shared__ float ipw[64];
  __shared__ float ipb[8];

  int t = threadIdx.x;
  int wave = t >> 5, lane = t & 31, lm = lane & 15, kh = lane >> 4;
  int tr = t >> 4, tc = t & 15;
  int R0 = blockIdx.x * 16;

  // Phase 0: per-row small vectors from GEMM1, zero att_w accumulator.
  for (int i = t; i < 1024; i += 256) {
    int r = i >> 6, c = i & 63;
    z1L[r][c] = z1g[(size_t)(R0 + r) * 64 + c];
    zfL[r][c] = zfg[(size_t)(R0 + r) * 64 + c];
    attw[r][c] = 0.f;
  }
  for (int i = t; i < 512; i += 256) {
    int r = i >> 5, c = i & 31;
    b1L[r][c]  = b1g[(size_t)(R0 + r) * 32 + c];
    v1L[r][c]  = v1g[(size_t)(R0 + r) * 32 + c];
    fa1L[r][c] = fa1g[(size_t)(R0 + r) * 32 + c];
  }
  __syncthreads();

  // Feature-weight softmax fw (B,4)
  if (tc < 4) {
    float s = fa_b2[tc];
    const float* wr = fa_w2 + tc * 32;
    for (int j = 0; j < 32; ++j) s += fa1L[tr][j] * wr[j];
    fwL[tr][tc] = s;
  }
  __syncthreads();
  if (tc == 0) {
    float m = fmaxf(fmaxf(fwL[tr][0], fwL[tr][1]), fmaxf(fwL[tr][2], fwL[tr][3]));
    float e0 = __expf(fwL[tr][0] - m), e1 = __expf(fwL[tr][1] - m);
    float e2 = __expf(fwL[tr][2] - m), e3 = __expf(fwL[tr][3] - m);
    float inv = 1.f / (e0 + e1 + e2 + e3);
    fwL[tr][0] = e0 * inv; fwL[tr][1] = e1 * inv;
    fwL[tr][2] = e2 * inv; fwL[tr][3] = e3 * inv;
  }
  __syncthreads();

  // Main loop over the 8 live (f,p) pairs.
  for (int fp = 0; fp < 8; ++fp) {
    int f = fp >> 1, p = fp & 1;
    if (p == 0) {  // stage feats[b,f,:,:] (16 x 512 f32)
      const float* fsrc = (f == 0) ? feats0 : (f == 1) ? feats1 : (f == 2) ? feats2 : feats3;
      for (int i = t; i < 2048; i += 256) {
        int r = i >> 7, c4 = (i & 127) * 4;
        *(float4*)&featL[r][c4] = *(const float4*)(fsrc + ((size_t)(R0 + r) * 512 + c4));
      }
    }
    if (t < 64) ipw[t] = inproj_w[(size_t)((f * 3 + p) * 8 + (t >> 3)) * 8 + (t & 7)];
    if (t < 8)  ipb[t] = inproj_b[(f * 3 + p) * 8 + t];
    __syncthreads();

    // GEMM2: W tile (16 rows x 512) = h_fp (16x64) @ hyp_w2^T, bf16 WMMA.
    const unsigned short* hrow = hbuf + (size_t)(R0 + lm) * 512 + fp * 64;
    for (int i = 0; i < 4; ++i) {           // 8 waves x 4 tiles = 32 N-tiles
      int n0 = (wave * 4 + i) * 16;
      v8f acc = {0.f, 0.f, 0.f, 0.f, 0.f, 0.f, 0.f, 0.f};
      const unsigned short* brow = wp2 + ((size_t)fp * 512 + n0 + lm) * 64;
#pragma unroll
      for (int kk = 0; kk < 2; ++kk) {
        int K0 = kk * 32;
        BFrag a, b;
        a.u[0] = *(const uint4*)(hrow + K0 + kh * 8);
        a.u[1] = *(const uint4*)(hrow + K0 + 16 + kh * 8);
        b.u[0] = *(const uint4*)(brow + K0 + kh * 8);
        b.u[1] = *(const uint4*)(brow + K0 + 16 + kh * 8);
        acc = __builtin_amdgcn_wmma_f32_16x16x32_bf16(false, a.v, false, b.v,
                                                      (short)0, acc, false, false);
      }
      float bias = hyp_b2[(size_t)(f * 3 + p) * 512 + n0 + lm];
#pragma unroll
      for (int r = 0; r < 8; ++r) Wt[r + 8 * kh][n0 + lm] = acc[r] + bias;
    }
    __syncthreads();

    // vec[r,a,e] = sum_c feats[r,a,c] * Wt[r, c*8+e]
    for (int j = 0; j < 4; ++j) {
      int ae = tc + 16 * j, aa = ae >> 3, e = ae & 7;
      const float* fr = &featL[tr][aa * 64];
      float s = 0.f;
      for (int c = 0; c < 64; ++c) s += fr[c] * Wt[tr][c * 8 + e];
      vecL[tr][ae] = s;
    }
    __syncthreads();

    // proj -> q (p==0) or k (p==1); HD^-0.5 == 1 since HD == 1.
    {
      float (*dst)[64] = (p == 0) ? qL : kL;
      for (int j = 0; j < 4; ++j) {
        int ad = tc + 16 * j, aa = ad >> 3, d = ad & 7;
        float s = ipb[d];
        for (int e = 0; e < 8; ++e) s += vecL[tr][aa * 8 + e] * ipw[d * 8 + e];
        dst[tr][aa * 8 + d] = s;
      }
    }
    __syncthreads();

    if (p == 1) {  // softmax attention (logits are outer products, HD=1), mean heads
      if (tc < 8) {
        int L = tc;
        float acc8[8] = {0, 0, 0, 0, 0, 0, 0, 0};
        for (int n = 0; n < 8; ++n) {
          float qv = qL[tr][L * 8 + n];
          float lg[8], mx = -3.4e38f;
          for (int s3 = 0; s3 < 8; ++s3) { lg[s3] = qv * kL[tr][s3 * 8 + n]; mx = fmaxf(mx, lg[s3]); }
          float sum = 0.f;
          for (int s3 = 0; s3 < 8; ++s3) { lg[s3] = __expf(lg[s3] - mx); sum += lg[s3]; }
          float inv = 1.f / sum;
          for (int s3 = 0; s3 < 8; ++s3) acc8[s3] += lg[s3] * inv;
        }
        float w = fwL[tr][f] * 0.125f;   // head mean folded in
        for (int s3 = 0; s3 < 8; ++s3) attw[tr][L * 8 + s3] += w * acc8[s3];
      }
      __syncthreads();
    }
  }

  // mixed = att_w @ qs
  if (tc < 8) {
    float s = 0.f;
    for (int j = 0; j < 8; ++j) s += attw[tr][tc * 8 + j] * agent_qs[(size_t)(R0 + tr) * 8 + j];
    mixedL[tr][tc] = s;
  }
  __syncthreads();

  // hidden[e] = elu(sum_a mixed[a]*|z1@hw1_b[a*32+e] + bb| + b1[e]); wf; y
  for (int j = 0; j < 2; ++j) {
    int e = tc + 16 * j;
    float s = b1L[tr][e];
    for (int a2 = 0; a2 < 8; ++a2) {
      int ae = a2 * 32 + e;
      const float* wr = hw1_b + (size_t)ae * 64;
      float d = hw1_bb[ae];
      for (int c = 0; c < 64; ++c) d += z1L[tr][c] * wr[c];
      s += mixedL[tr][a2] * fabsf(d);
    }
    hidL[tr][e] = (s > 0.f) ? s : (__expf(s) - 1.f);   // elu
    const float* wr = hwf_b + (size_t)e * 64;
    float w = hwf_bb[e];
    for (int c = 0; c < 64; ++c) w += zfL[tr][c] * wr[c];
    wfL[tr][e] = fabsf(w);
  }
  __syncthreads();

  if (tc == 0) {
    float v = v2_b[0];
    for (int j = 0; j < 32; ++j) v += v1L[tr][j] * v2_w[j];
    float y = v;
    for (int e = 0; e < 32; ++e) y += hidL[tr][e] * wfL[tr][e];
    out[R0 + tr] = y;
  }
}

// ---------------------------------------------------------------------------
extern "C" void kernel_launch(void* const* d_in, const int* in_sizes, int n_in,
                              void* d_out, int out_size, void* d_ws, size_t ws_size,
                              hipStream_t stream) {
  (void)in_sizes; (void)n_in; (void)out_size; (void)ws_size;
  const float* agent_qs = (const float*)d_in[0];
  const float* states   = (const float*)d_in[1];
  const float* move_f   = (const float*)d_in[2];
  const float* enemy_f  = (const float*)d_in[3];
  const float* ally_f   = (const float*)d_in[4];
  const float* own_f    = (const float*)d_in[5];
  const float* hyp_w1   = (const float*)d_in[6];
  const float* hyp_b1   = (const float*)d_in[7];
  const float* hyp_w2   = (const float*)d_in[8];
  const float* hyp_b2   = (const float*)d_in[9];
  const float* inproj_w = (const float*)d_in[10];
  const float* inproj_b = (const float*)d_in[11];
  const float* fa_w1    = (const float*)d_in[12];
  const float* fa_b1    = (const float*)d_in[13];
  const float* fa_w2    = (const float*)d_in[14];
  const float* fa_b2    = (const float*)d_in[15];
  const float* hw1_a    = (const float*)d_in[16];
  const float* hw1_ab   = (const float*)d_in[17];
  const float* hw1_b    = (const float*)d_in[18];
  const float* hw1_bb   = (const float*)d_in[19];
  const float* hwf_a    = (const float*)d_in[20];
  const float* hwf_ab   = (const float*)d_in[21];
  const float* hwf_b    = (const float*)d_in[22];
  const float* hwf_bb   = (const float*)d_in[23];
  const float* hb1_w    = (const float*)d_in[24];
  const float* hb1_b    = (const float*)d_in[25];
  const float* v1_w     = (const float*)d_in[26];
  const float* v1_b     = (const float*)d_in[27];
  const float* v2_w     = (const float*)d_in[28];
  const float* v2_b     = (const float*)d_in[29];

  char* ws = (char*)d_ws;
  size_t off = 0;
  auto take = [&](size_t bytes) {
    char* p = ws + off;
    off += (bytes + 255) & ~(size_t)255;
    return p;
  };
  unsigned short* stbf = (unsigned short*)take((size_t)BATCH * 256 * 2);  //  8 MB
  unsigned short* wp1  = (unsigned short*)take((size_t)N1 * 256 * 2);     // 384 KB
  float*          bp1  = (float*)take((size_t)N1 * 4);
  unsigned short* wp2  = (unsigned short*)take((size_t)8 * 512 * 64 * 2); // 512 KB
  unsigned short* hbuf = (unsigned short*)take((size_t)BATCH * 512 * 2);  // 16 MB
  float* z1g  = (float*)take((size_t)BATCH * 64 * 4);
  float* zfg  = (float*)take((size_t)BATCH * 64 * 4);
  float* b1g  = (float*)take((size_t)BATCH * 32 * 4);
  float* v1g  = (float*)take((size_t)BATCH * 32 * 4);
  float* fa1g = (float*)take((size_t)BATCH * 32 * 4);    // total ~41 MB workspace

  k_cvt_states<<<(BATCH * 256) / (256 * 4), 256, 0, stream>>>(states, stbf);
  k_pack_w1<<<N1, 256, 0, stream>>>(hyp_w1, hyp_b1, hw1_a, hw1_ab, hwf_a, hwf_ab,
                                    hb1_w, hb1_b, v1_w, v1_b, fa_w1, fa_b1, wp1, bp1);
  k_pack_w2<<<(8 * 512 * 64) / 256, 256, 0, stream>>>(hyp_w2, wp2);

  dim3 g1(BATCH / 16, N1 / 128);
  k_gemm1<<<g1, 256, 0, stream>>>(stbf, wp1, bp1, hbuf, z1g, zfg, b1g, v1g, fa1g);

  k_attn_mix_tail<<<BATCH / 16, 256, 0, stream>>>(
      hbuf, wp2, hyp_b2, inproj_w, inproj_b, move_f, enemy_f, ally_f, own_f,
      agent_qs, fa_w2, fa_b2, hw1_b, hw1_bb, hwf_b, hwf_bb, v2_w, v2_b,
      z1g, zfg, b1g, v1g, fa1g, (float*)d_out);
}
